// net_crossInteraction_77704548319766
// MI455X (gfx1250) — matrix operand order repaired and checked by reading, not compiled, not run on hardware
//
#include <hip/hip_runtime.h>
#include <math.h>

// ---------------------------------------------------------------------------
// Sizes (fixed by the reference): b=8 per branch, combined batch B2=16,
// n = 16*16 = 256 nodes, d = 128, GRU T = 16, 3d = 384.
// ---------------------------------------------------------------------------
#define DV   128
#define NN   256
#define BB   8
#define B2   16
#define TT   16
#define H3   384

typedef __attribute__((ext_vector_type(16))) _Float16 v16h;
typedef __attribute__((ext_vector_type(8)))  float    v8f;

union AF { v16h v; _Float16 h[16]; };
union CF { v8f  v; float    f[8];  };

__device__ __forceinline__ float sigmoidf_(float x) { return 1.f / (1.f + expf(-x)); }
__device__ __forceinline__ float leakyf_(float x)   { return (x >= 0.f) ? x : 0.1f * x; }

// A-fragment (16x32 f16) contiguous k order, k = 16*half + s (B matches); converts f32 -> f16 on load.
__device__ __forceinline__ AF load_a_frag(const float* __restrict__ ar, int half)
{
    AF a;
    #pragma unroll
    for (int v = 0; v < 8; ++v) {
        int ka = (half << 4) + (v << 1);
        a.h[2*v]   = (_Float16)ar[ka];
        a.h[2*v+1] = (_Float16)ar[ka + 1];
    }
    return a;
}

// B-fragment (32x16 f16) from weight rows: B[k,n] = W[n*ldw + k]
__device__ __forceinline__ AF load_b_frag_wt(const float* __restrict__ br, int half)
{
    AF b;
    #pragma unroll
    for (int v = 0; v < 8; ++v) {
        int kb = (half << 4) + (v << 1);
        b.h[2*v]   = (_Float16)br[kb];
        b.h[2*v+1] = (_Float16)br[kb + 1];
    }
    return b;
}

// B-fragment from a plain K x N row-major matrix: B[k,n] = Bm[k*ldb + n]
__device__ __forceinline__ AF load_b_frag_n(const float* __restrict__ bp, int ldb, int half)
{
    AF b;
    #pragma unroll
    for (int v = 0; v < 8; ++v) {
        int kb = (half << 4) + (v << 1);
        b.h[2*v]   = (_Float16)bp[(long long)kb * ldb];
        b.h[2*v+1] = (_Float16)bp[(long long)(kb + 1) * ldb];
    }
    return b;
}

// ---------------------------------------------------------------------------
// Generic batched GEMM. One wave (32 threads) per 16x64 C strip: 4 WMMA
// accumulators sharing one A fragment per K-chunk (4x A-traffic reduction).
//   C = act( A @ op(B) + bias ) (+ residual)
//   wt==1 : B element [k,n] = B[n*ldb + k]  (weight stored N x K row-major)
//   wt==0 : B element [k,n] = B[k*ldb + n]  (plain K x N row-major)
//   act: 0 none, 1 relu, 2 sigmoid
// Requires: M % 16 == 0, N % 64 == 0, K % 32 == 0 (true for every call here).
// ---------------------------------------------------------------------------
__global__ __launch_bounds__(32)
void k_gemm(const float* __restrict__ A, long long sA,
            const float* __restrict__ B, long long sB, int ldb, int wt,
            const float* __restrict__ bias,
            const float* __restrict__ R,
            float* __restrict__ C, long long sC,
            int M, int N, int K, int lda, int ldc, int act)
{
    (void)M; (void)N;
    const int lane = threadIdx.x;
    const int tn = blockIdx.x << 6;   // 64-wide N strip
    const int tm = blockIdx.y << 4;
    const int m_ = lane & 15;         // row (A) / col (B,C) within tile
    const int half = lane >> 4;

    const float* Ab = A + (long long)blockIdx.z * sA + (long long)tm * lda;
    const float* Bb = B + (long long)blockIdx.z * sB;

    CF acc[4];
    #pragma unroll
    for (int j = 0; j < 4; ++j) acc[j].v = (v8f){0.f,0.f,0.f,0.f,0.f,0.f,0.f,0.f};

    for (int k0 = 0; k0 < K; k0 += 32) {
        AF a = load_a_frag(Ab + (long long)m_ * lda + k0, half);
        if (wt) {
            #pragma unroll
            for (int j = 0; j < 4; ++j) {
                AF b = load_b_frag_wt(Bb + (long long)(tn + (j << 4) + m_) * ldb + k0, half);
                acc[j].v = __builtin_amdgcn_wmma_f32_16x16x32_f16(
                    false, a.v, false, b.v, (short)0, acc[j].v, false, false);
            }
        } else {
            #pragma unroll
            for (int j = 0; j < 4; ++j) {
                AF b = load_b_frag_n(Bb + (long long)k0 * ldb + tn + (j << 4) + m_, ldb, half);
                acc[j].v = __builtin_amdgcn_wmma_f32_16x16x32_f16(
                    false, a.v, false, b.v, (short)0, acc[j].v, false, false);
            }
        }
    }

    const int n = m_;
    const int mb = half << 3;
    float* Cb = C + (long long)blockIdx.z * sC;
    const float* Rb = R ? R + (long long)blockIdx.z * sC : nullptr;
    #pragma unroll
    for (int j = 0; j < 4; ++j) {
        const int nc = tn + (j << 4) + n;
        const float bv = bias ? bias[nc] : 0.f;
        #pragma unroll
        for (int v = 0; v < 8; ++v) {
            int m = tm + mb + v;
            float x = acc[j].f[v] + bv;
            if (act == 1)      x = fmaxf(x, 0.f);
            else if (act == 2) x = sigmoidf_(x);
            if (Rb) x += Rb[(long long)m * ldc + nc];
            Cb[(long long)m * ldc + nc] = x;
        }
    }
}

// ---------------------------------------------------------------------------
// Embedding gather into the combined batch: slots 0..7 = prot_data2 (branch1),
// slots 8..15 = prot_data1 (branch2).
// ---------------------------------------------------------------------------
__global__ void k_embed(const int* __restrict__ p1, const int* __restrict__ p2,
                        const float* __restrict__ emb, float* __restrict__ e)
{
    int idx = blockIdx.x * blockDim.x + threadIdx.x;
    if (idx >= B2 * NN * DV) return;
    int d = idx & (DV - 1);
    int p = (idx >> 7) & (NN - 1);
    int z = idx >> 15;
    const int* tok = (z < BB) ? (p2 + z * NN) : (p1 + (z - BB) * NN);
    e[idx] = emb[tok[p] * DV + d];
}

// ---------------------------------------------------------------------------
// Fused GRU scan: block = 16 rows, 128 threads (4 waves). Per time step:
// hw = h @ whh^T + bhh via WMMA. Each wave owns a 16x96 strip (6 tiles):
// the LDS h-fragment is loaded once per K-chunk and reused across 6 WMMAs.
// h and hw live in LDS for the entire recurrence.
// ---------------------------------------------------------------------------
__global__ __launch_bounds__(128)
void k_gru_scan(const float* __restrict__ xw,   // [rows, T, 384]
                const float* __restrict__ whh,  // [384, 128]
                const float* __restrict__ bhh,  // [384]
                float* __restrict__ out)        // [rows, T, 128]
{
    __shared__ float h[16][DV];
    __shared__ float hw[16][H3];
    const int tid = threadIdx.x;
    const int lane = tid & 31;
    const int wv = tid >> 5;
    const int rowbase = blockIdx.x * 16;
    const int m_ = lane & 15, half = lane >> 4;

    #pragma unroll
    for (int e = 0; e < 16; ++e) {
        int idx = tid + (e << 7);
        h[idx >> 7][idx & (DV - 1)] = 0.f;
    }
    __syncthreads();

    for (int t = 0; t < TT; ++t) {
        CF acc[6];
        #pragma unroll
        for (int j = 0; j < 6; ++j) acc[j].v = (v8f){0.f,0.f,0.f,0.f,0.f,0.f,0.f,0.f};

        for (int k0 = 0; k0 < DV; k0 += 32) {
            AF a = load_a_frag(&h[m_][k0], half);      // from LDS
            #pragma unroll
            for (int j = 0; j < 6; ++j) {
                int tn = (wv * 6 + j) << 4;
                AF b = load_b_frag_wt(whh + (long long)(tn + m_) * DV + k0, half);
                acc[j].v = __builtin_amdgcn_wmma_f32_16x16x32_f16(
                    false, a.v, false, b.v, (short)0, acc[j].v, false, false);
            }
        }
        {
            int n = m_, mb = half << 3;
            #pragma unroll
            for (int j = 0; j < 6; ++j) {
                int tn = (wv * 6 + j) << 4;
                #pragma unroll
                for (int v = 0; v < 8; ++v)
                    hw[mb + v][tn + n] = acc[j].f[v] + bhh[tn + n];
            }
        }
        __syncthreads();
        #pragma unroll
        for (int e = 0; e < 16; ++e) {
            int idx = tid + (e << 7);
            int r = idx >> 7, c = idx & (DV - 1);
            const float* xr = xw + ((long long)(rowbase + r) * TT + t) * H3;
            float rg = sigmoidf_(xr[c]          + hw[r][c]);
            float zg = sigmoidf_(xr[DV + c]     + hw[r][DV + c]);
            float cg = tanhf   (xr[2 * DV + c]  + rg * hw[r][2 * DV + c]);
            float hn = (1.f - zg) * cg + zg * h[r][c];
            h[r][c] = hn;
            out[((long long)(rowbase + r) * TT + t) * DV + c] = hn;
        }
        __syncthreads();
    }
}

// ---------------------------------------------------------------------------
// GAT attention post: a = sigmoid(logit) + 1e-5*eye; a *= adj (diag forced 1);
// row-normalize. One block per (batch, row).
// ---------------------------------------------------------------------------
__global__ __launch_bounds__(256)
void k_attn_post(float* __restrict__ a, const float* __restrict__ adj1,
                 const float* __restrict__ adj2)
{
    __shared__ float red[256];
    const int row = blockIdx.x, z = blockIdx.y, col = threadIdx.x;
    const long long base = ((long long)z * NN + row) * NN;
    float v = sigmoidf_(a[base + col]);
    if (row == col) v += 1e-5f;
    float adjv;
    if (row == col) adjv = 1.f;
    else {
        const float* adj = (z < BB) ? (adj1 + (long long)z * NN * NN)
                                    : (adj2 + (long long)(z - BB) * NN * NN);
        adjv = adj[(long long)row * NN + col];
    }
    v *= adjv;
    red[col] = v;
    __syncthreads();
    for (int o = 128; o > 0; o >>= 1) { if (col < o) red[col] += red[col + o]; __syncthreads(); }
    a[base + col] = v / red[0];
}

// ---------------------------------------------------------------------------
// Cross-interaction gates: ci0 = tanh(<m0, s>)+1, ci1 = tanh(<m1, g>)+1
// ---------------------------------------------------------------------------
__global__ __launch_bounds__(128)
void k_cross_ci(const float* __restrict__ m0, const float* __restrict__ s,
                const float* __restrict__ m1, const float* __restrict__ g,
                float* __restrict__ ci0, float* __restrict__ ci1)
{
    __shared__ float red[128];
    const int bn = blockIdx.x, d = threadIdx.x;
    const long long base = (long long)bn * DV + d;
    red[d] = m0[base] * s[base];
    __syncthreads();
    for (int o = 64; o > 0; o >>= 1) { if (d < o) red[d] += red[d + o]; __syncthreads(); }
    if (d == 0) ci0[bn] = tanhf(red[0]) + 1.f;
    __syncthreads();
    red[d] = m1[base] * g[base];
    __syncthreads();
    for (int o = 64; o > 0; o >>= 1) { if (d < o) red[d] += red[d + o]; __syncthreads(); }
    if (d == 0) ci1[bn] = tanhf(red[0]) + 1.f;
}

__global__ void k_cat(const float* __restrict__ s, const float* __restrict__ g,
                      const float* __restrict__ ci0, const float* __restrict__ ci1,
                      float* __restrict__ cat)
{
    int idx = blockIdx.x * blockDim.x + threadIdx.x;
    if (idx >= B2 * NN * 2 * DV) return;
    int c = idx & 255, bn = idx >> 8;
    cat[idx] = (c < DV) ? s[(long long)bn * DV + c] * ci0[bn]
                        : g[(long long)bn * DV + (c - DV)] * ci1[bn];
}

__global__ void k_relu(const float* __restrict__ in, float* __restrict__ out, int n)
{
    int i = blockIdx.x * blockDim.x + threadIdx.x;
    if (i < n) out[i] = fmaxf(in[i], 0.f);
}

__global__ void k_zero(float* __restrict__ p, int n)
{
    int i = blockIdx.x * blockDim.x + threadIdx.x;
    if (i < n) p[i] = 0.f;
}

__global__ __launch_bounds__(256)
void k_len(const int* __restrict__ p1, const int* __restrict__ p2, int* __restrict__ lens)
{
    __shared__ int red[256];
    const int b = blockIdx.x, which = blockIdx.y, t = threadIdx.x;
    const int* tok = (which == 0) ? (p1 + b * NN) : (p2 + b * NN);
    red[t] = (tok[t] != 0) ? 1 : 0;
    __syncthreads();
    for (int o = 128; o > 0; o >>= 1) { if (t < o) red[t] += red[t + o]; __syncthreads(); }
    if (t == 0) lens[which * BB + b] = red[0];   // lens[0..7]=len1, [8..15]=len2
}

__global__ void k_mask(const float* __restrict__ inter, const int* __restrict__ lens,
                       float* __restrict__ out)
{
    int idx = blockIdx.x * blockDim.x + threadIdx.x;
    if (idx >= BB * NN * NN) return;
    int k = idx & 255, i = (idx >> 8) & 255, b = idx >> 16;
    float m = ((i < lens[BB + b]) ? 1.f : 0.f) * ((k < lens[b]) ? 1.f : 0.f);
    out[idx] = inter[idx] * m;
}

// pp[b,d] = sum_{i,k} tanh(pe1[b,i,d]*pe2[b,k,d]) * inter[b,i,k]
__global__ __launch_bounds__(128)
void k_pp(const float* __restrict__ pe, const float* __restrict__ inter,
          float* __restrict__ pp)
{
    const int i = blockIdx.x, b = blockIdx.y, d = threadIdx.x;
    const float* pe1 = pe + ((long long)b * NN + i) * DV;
    const float* pe2 = pe + ((long long)(BB + b) * NN) * DV;
    const float* irow = inter + ((long long)b * NN + i) * NN;
    const float p1 = pe1[d];
    float acc = 0.f;
    for (int k = 0; k < NN; ++k)
        acc += tanhf(p1 * pe2[(long long)k * DV + d]) * irow[k];
    atomicAdd(&pp[b * DV + d], acc);
}

// conv1d(k=4, stride=2, pad=1) + leaky(0.1) + maxpool over groups of 4 positions
__global__ __launch_bounds__(1024)
void k_conv(const float* __restrict__ pp, const float* __restrict__ cw,
            const float* __restrict__ cb, float* __restrict__ feat)
{
    const int b = blockIdx.x, t = threadIdx.x;
    const int ch = t >> 4, g = t & 15;
    float m = -1e30f;
    for (int q = 0; q < 4; ++q) {
        int p = g * 4 + q;
        float acc = cb[ch];
        #pragma unroll
        for (int k = 0; k < 4; ++k) {
            int pos = p * 2 - 1 + k;
            float v = (pos >= 0 && pos < DV) ? pp[b * DV + pos] : 0.f;
            acc += cw[ch * 4 + k] * v;
        }
        m = fmaxf(m, leakyf_(acc));
    }
    feat[b * 1024 + ch * 16 + g] = m;   // matches reshape(b,64,16)
}

__global__ void k_dense(const float* __restrict__ in, const float* __restrict__ w,
                        const float* __restrict__ bias, float* __restrict__ out,
                        int B, int IN, int OUT, int act)
{
    int idx = blockIdx.x * blockDim.x + threadIdx.x;
    if (idx >= B * OUT) return;
    int b = idx / OUT, o = idx % OUT;
    float acc = bias[o];
    const float* ip = in + (long long)b * IN;
    const float* wp = w + (long long)o * IN;
    for (int i = 0; i < IN; ++i) acc += ip[i] * wp[i];
    if (act == 1) acc = leakyf_(acc);
    out[idx] = acc;
}

// ---------------------------------------------------------------------------
// Host-side orchestration
// ---------------------------------------------------------------------------
extern "C" void kernel_launch(void* const* d_in, const int* in_sizes, int n_in,
                              void* d_out, int out_size, void* d_ws, size_t ws_size,
                              hipStream_t stream)
{
    (void)in_sizes; (void)n_in; (void)out_size;
    const int*   prot1 = (const int*)  d_in[0];
    const int*   prot2 = (const int*)  d_in[1];
    const float* adj1  = (const float*)d_in[2];
    const float* adj2  = (const float*)d_in[3];
    const float* emb   = (const float*)d_in[4];
    const float* g0wih = (const float*)d_in[5];
    const float* g0whh = (const float*)d_in[6];
    const float* g0bih = (const float*)d_in[7];
    const float* g0bhh = (const float*)d_in[8];
    const float* g1wih = (const float*)d_in[9];
    const float* g1whh = (const float*)d_in[10];
    const float* g1bih = (const float*)d_in[11];
    const float* g1bhh = (const float*)d_in[12];
    const float* gaww  = (const float*)d_in[13];
    const float* gawb  = (const float*)d_in[14];
    const float* gal0w = (const float*)d_in[15];
    const float* gal0b = (const float*)d_in[16];
    const float* gal1w = (const float*)d_in[17];
    const float* gal1b = (const float*)d_in[18];
    const float* gatfw = (const float*)d_in[19];
    const float* gatfb = (const float*)d_in[20];
    const float* c0w1  = (const float*)d_in[21];
    const float* c0b1  = (const float*)d_in[22];
    const float* c0w2  = (const float*)d_in[23];
    const float* c0b2  = (const float*)d_in[24];
    const float* c1w1  = (const float*)d_in[25];
    const float* c1b1  = (const float*)d_in[26];
    const float* c1w2  = (const float*)d_in[27];
    const float* c1b2  = (const float*)d_in[28];
    const float* cilw  = (const float*)d_in[29];
    const float* cilb  = (const float*)d_in[30];
    const float* ja1w  = (const float*)d_in[31];
    const float* ja1b  = (const float*)d_in[32];
    const float* ja2w  = (const float*)d_in[33];
    const float* ja2b  = (const float*)d_in[34];
    const float* convw = (const float*)d_in[35];
    const float* convb = (const float*)d_in[36];
    const float* r1w   = (const float*)d_in[37];
    const float* r1b   = (const float*)d_in[38];
    const float* r2w   = (const float*)d_in[39];
    const float* r2b   = (const float*)d_in[40];
    const float* r3w   = (const float*)d_in[41];
    const float* r3b   = (const float*)d_in[42];

    // Workspace arena (f32). Deterministic layout recomputed each call.
    float* base = (float*)d_ws;
    size_t off = 0;
    auto A_ = [&](size_t n) { float* p = base + off; off += n; return p; };
    float* b_e   = A_(524288);    // combined embeddings / GAT layer-1 input
    float* b_xw  = A_(1572864);   // GRU xw precompute (4096 x 384)
    float* b_s   = A_(524288);    // GRU output (x_seq)
    float* b_q   = A_(524288);    // GAT q / MLP hidden
    float* b_log = A_(1048576);   // GAT attention logits; reused as cross-cat
    float* b_t   = A_(524288);    // GAT aggregation buffer
    float* b_xA  = A_(524288);
    float* b_xB  = A_(524288);
    float* b_g   = A_(524288);    // GAT output (x_graph)
    float* b_m0  = A_(524288);
    float* b_m1  = A_(524288);
    float* b_ci  = A_(8192);      // ci0 @ +0, ci1 @ +4096
    float* b_pe  = A_(524288);
    float* b_rp  = A_(524288);    // relu(pe)
    float* b_u   = A_(524288);    // u1 @ +0 (8 batches), u2 @ +262144
    float* b_int = A_(524288);    // interaction map (unmasked)
    float* b_pp  = A_(1024);
    float* b_ft  = A_(8192);
    float* b_f1  = A_(4800);
    float* b_f2  = A_(2400);
    int*   b_ln  = (int*)A_(16);
    if (off * sizeof(float) > ws_size) return;   // ws too small: bail

    auto gemm = [&](const float* A, long long sA, const float* Bm, long long sB,
                    int ldb, int wt, const float* bias, const float* R,
                    float* C, long long sC, int M, int N, int K,
                    int lda, int ldc, int act, int batch) {
        dim3 grid(N / 64, M / 16, batch);
        k_gemm<<<grid, dim3(32), 0, stream>>>(A, sA, Bm, sB, ldb, wt, bias, R,
                                              C, sC, M, N, K, lda, ldc, act);
    };

    // 1) Embedding (both branches into combined batch of 16)
    k_embed<<<(B2 * NN * DV + 255) / 256, 256, 0, stream>>>(prot1, prot2, emb, b_e);

    // 2) GRU0: xw = e @ wih^T + bih (M=4096,K=128,N=384), then fused scan
    gemm(b_e, 0, g0wih, 0, DV, 1, g0bih, nullptr, b_xw, 0, B2 * NN, H3, DV, DV, H3, 0, 1);
    k_gru_scan<<<16, 128, 0, stream>>>(b_xw, g0whh, g0bhh, b_s);
    // 3) GRU1 (reshape is a pure reinterpretation of the flat buffer)
    gemm(b_s, 0, g1wih, 0, DV, 1, g1bih, nullptr, b_xw, 0, B2 * NN, H3, DV, DV, H3, 0, 1);
    k_gru_scan<<<16, 128, 0, stream>>>(b_xw, g1whh, g1bhh, b_s);

    // 4) GAT: 7 layers, combined batch 16
    const float* xin = b_e;
    for (int l = 0; l < 7; ++l) {
        const float* ww = gaww  + (long long)l * DV * DV;
        const float* wb = gawb  + (long long)l * DV;
        const float* w0 = gal0w + (long long)l * DV * DV;
        const float* b0 = gal0b + (long long)l * DV;
        const float* w1 = gal1w + (long long)l * DV * DV;
        const float* b1 = gal1b + (long long)l * DV;
        // q = x @ ww^T + wb (flattened over batch)
        gemm(xin, 0, ww, 0, DV, 1, wb, nullptr, b_q, 0, B2 * NN, DV, DV, DV, DV, 0, 1);
        // logits = q @ x^T (batched, B as "weight" = x row-major)
        gemm(b_q, (long long)NN * DV, xin, (long long)NN * DV, DV, 1, nullptr, nullptr,
             b_log, (long long)NN * NN, NN, NN, DV, DV, NN, 0, B2);
        k_attn_post<<<dim3(NN, B2), 256, 0, stream>>>(b_log, adj1, adj2);
        // x1 = a @ x (batched, plain K x N B)
        gemm(b_log, (long long)NN * NN, xin, (long long)NN * DV, DV, 0, nullptr, nullptr,
             b_t, (long long)NN * DV, NN, DV, NN, NN, DV, 0, B2);
        // h = relu(x1 @ w0^T + b0)
        gemm(b_t, 0, w0, 0, DV, 1, b0, nullptr, b_q, 0, B2 * NN, DV, DV, DV, DV, 1, 1);
        // x = relu(h @ w1^T + b1) + x0
        float* xout = (l % 2 == 0) ? b_xA : b_xB;
        gemm(b_q, 0, w1, 0, DV, 1, b1, xin, xout, 0, B2 * NN, DV, DV, DV, DV, 1, 1);
        xin = xout;
    }
    // g = x @ fw^T + fb
    gemm(xin, 0, gatfw, 0, DV, 1, gatfb, nullptr, b_g, 0, B2 * NN, DV, DV, DV, DV, 0, 1);

    // 5) Cross interaction
    gemm(b_g, 0, c0w1, 0, DV, 1, c0b1, nullptr, b_q,  0, B2 * NN, DV, DV, DV, DV, 1, 1);
    gemm(b_q, 0, c0w2, 0, DV, 1, c0b2, nullptr, b_m0, 0, B2 * NN, DV, DV, DV, DV, 0, 1);
    gemm(b_s, 0, c1w1, 0, DV, 1, c1b1, nullptr, b_q,  0, B2 * NN, DV, DV, DV, DV, 1, 1);
    gemm(b_q, 0, c1w2, 0, DV, 1, c1b2, nullptr, b_m1, 0, B2 * NN, DV, DV, DV, DV, 0, 1);
    k_cross_ci<<<B2 * NN, 128, 0, stream>>>(b_m0, b_s, b_m1, b_g, b_ci, b_ci + 4096);
    k_cat<<<(B2 * NN * 2 * DV + 255) / 256, 256, 0, stream>>>(b_s, b_g, b_ci, b_ci + 4096, b_log);
    // pe = cat @ lw^T + lb (K = 256)
    gemm(b_log, 0, cilw, 0, 2 * DV, 1, cilb, nullptr, b_pe, 0, B2 * NN, DV, 2 * DV, 2 * DV, DV, 0, 1);

    // 6) Interaction map: sigmoid( (relu(pe1)@ja2^T) @ (relu(pe2)@ja1^T)^T )
    k_relu<<<(B2 * NN * DV + 255) / 256, 256, 0, stream>>>(b_pe, b_rp, B2 * NN * DV);
    gemm(b_rp,               0, ja2w, 0, DV, 1, ja2b, nullptr, b_u,            0, BB * NN, DV, DV, DV, DV, 0, 1);
    gemm(b_rp + BB * NN * DV, 0, ja1w, 0, DV, 1, ja1b, nullptr, b_u + BB*NN*DV, 0, BB * NN, DV, DV, DV, DV, 0, 1);
    gemm(b_u, (long long)NN * DV, b_u + BB * NN * DV, (long long)NN * DV, DV, 1,
         nullptr, nullptr, b_int, (long long)NN * NN, NN, NN, DV, DV, NN, 2, BB);

    // 7) Length masks -> d_out[0 .. 8*256*256)
    k_len<<<dim3(BB, 2), 256, 0, stream>>>(prot1, prot2, b_ln);
    k_mask<<<(BB * NN * NN + 255) / 256, 256, 0, stream>>>(b_int, b_ln, (float*)d_out);

    // 8) pp reduction (uses unmasked inter), then conv + maxpool + MLP head
    k_zero<<<(BB * DV + 255) / 256, 256, 0, stream>>>(b_pp, BB * DV);
    k_pp<<<dim3(NN, BB), 128, 0, stream>>>(b_pe, b_int, b_pp);
    k_conv<<<BB, 1024, 0, stream>>>(b_pp, convw, convb, b_ft);
    k_dense<<<(BB * 600 + 255) / 256, 256, 0, stream>>>(b_ft, r1w, r1b, b_f1, BB, 1024, 600, 1);
    k_dense<<<(BB * 300 + 255) / 256, 256, 0, stream>>>(b_f1, r2w, r2b, b_f2, BB, 600, 300, 1);
    k_dense<<<1, 256, 0, stream>>>(b_f2, r3w, r3b, (float*)d_out + BB * NN * NN, BB, 300, 1, 0);
}
